// FixedLLaMAMoE_7017976561987
// MI455X (gfx1250) — matrix operand order repaired and checked by reading, not compiled, not run on hardware
//
#include <hip/hip_runtime.h>
#include <hip/hip_bf16.h>
#include <math.h>

// ---------------- problem constants (match reference setup_inputs) ----------
#define NTOK 4096        // B*T = 2*2048 tokens
#define CDIM 2048        // model dim C
#define EXP  8           // experts
#define FDIM 1024        // ffn dim F
#define NSLOT (2*NTOK)   // top-2 -> 8192 (token,expert) slots

// ---------------- GEMM tiling ----------------
#define BM 128           // token rows per block
#define BN 64            // output cols per block
#define BK 32            // K per iteration == WMMA K for bf16
#define LDK 40           // padded ushorts per LDS row (80B, 16B aligned, bank-spread)

typedef __attribute__((ext_vector_type(16))) __bf16 v16bf;
typedef __attribute__((ext_vector_type(8)))  float  v8f;

struct alignas(16) U4 { unsigned int x, y, z, w; };
union FragU { v16bf v; U4 q[2]; };

// fp32 -> bf16 round-to-nearest-even
__device__ __forceinline__ unsigned short f2bf(float f) {
  unsigned u = __float_as_uint(f);
  unsigned r = (u + 0x7FFFu + ((u >> 16) & 1u)) >> 16;
  return (unsigned short)r;
}

__device__ __forceinline__ v8f wmma_bf16(v16bf a, v16bf b, v8f c) {
  // v_wmma_f32_16x16x32_bf16: (neg_a, A, neg_b, B, c_mod, C, reuse_a, reuse_b)
  return __builtin_amdgcn_wmma_f32_16x16x32_bf16(false, a, false, b, (short)0, c,
                                                 false, false);
}

// A fragment (16x32 bf16, M across lanes%16): ISA 7.12.2 16-bit A layout.
// lanes<16: elems 0..7 = K0..7,   8..15 = K16..23
// lanes>=16: elems 0..7 = K8..15, 8..15 = K24..31
__device__ __forceinline__ v16bf ld_fragA(const unsigned short* tile, int row, int lane) {
  int m  = row + (lane & 15);
  int kh = (lane & 16) ? 8 : 0;
  const unsigned short* p = tile + m * LDK + kh;
  FragU f;
  f.q[0] = *(const U4*)(p);
  f.q[1] = *(const U4*)(p + 16);
  return f.v;
}

// B fragment (32x16 bf16, N across lanes%16):
// lanes<16: elems 0..7 = K0..7, 8..15 = K16..23; lanes>=16: K8..15, K24..31
__device__ __forceinline__ v16bf ld_fragB(const unsigned short* tile, int row, int lane) {
  int n  = row + (lane & 15);
  int ko = (lane & 16) ? 8 : 0;
  const unsigned short* p = tile + n * LDK + ko;
  FragU f;
  f.q[0] = *(const U4*)(p);
  f.q[1] = *(const U4*)(p + 16);
  return f.v;
}

// ---------------- conversion kernels ----------------
__global__ void cvt_x_kernel(const float* __restrict__ x,
                             unsigned short* __restrict__ xb, int n) {
  int i = blockIdx.x * blockDim.x + threadIdx.x;
  if (i < n) xb[i] = f2bf(x[i]);
}

// per-expert transpose+convert: src [z][rows][cols] f32 -> dst [z][cols][rows] bf16
__global__ void transpose_bf16_kernel(const float* __restrict__ src,
                                      unsigned short* __restrict__ dst,
                                      int rows, int cols) {
  __shared__ float tile[32][33];
  size_t eoff = (size_t)blockIdx.z * rows * cols;
  src += eoff; dst += eoff;
  int c0 = blockIdx.x * 32, r0 = blockIdx.y * 32;
  int tx = threadIdx.x, ty = threadIdx.y;     // 32 x 8
#pragma unroll
  for (int i = 0; i < 32; i += 8)
    tile[ty + i][tx] = src[(size_t)(r0 + ty + i) * cols + c0 + tx];
  __syncthreads();
#pragma unroll
  for (int i = 0; i < 32; i += 8)
    dst[(size_t)(c0 + ty + i) * rows + r0 + tx] = f2bf(tile[tx][ty + i]);
}

// ---------------- router: fp32 logits, top-2, softmax ----------------
__global__ void router_kernel(const float* __restrict__ x,
                              const float* __restrict__ wr,
                              int* __restrict__ counts,
                              int* __restrict__ tidx,
                              float* __restrict__ tp) {
  int lane = threadIdx.x & 31;
  int wave = threadIdx.x >> 5;
  int n = blockIdx.x * 8 + wave;              // one wave per token
  const float* xr = x + (size_t)n * CDIM;
  float acc[EXP];
#pragma unroll
  for (int e = 0; e < EXP; e++) acc[e] = 0.f;
  for (int c = lane; c < CDIM; c += 32) {
    float xv = xr[c];
    const float4* w4 = (const float4*)(wr + (size_t)c * EXP);
    float4 wa = w4[0], wb = w4[1];
    acc[0] += xv * wa.x; acc[1] += xv * wa.y; acc[2] += xv * wa.z; acc[3] += xv * wa.w;
    acc[4] += xv * wb.x; acc[5] += xv * wb.y; acc[6] += xv * wb.z; acc[7] += xv * wb.w;
  }
#pragma unroll
  for (int off = 16; off >= 1; off >>= 1) {
#pragma unroll
    for (int e = 0; e < EXP; e++) acc[e] += __shfl_xor(acc[e], off, 32);
  }
  if (lane == 0) {
    int e0 = 0; float v0 = acc[0];
#pragma unroll
    for (int e = 1; e < EXP; e++) if (acc[e] > v0) { v0 = acc[e]; e0 = e; }
    int e1 = -1; float v1 = -3.4e38f;
#pragma unroll
    for (int e = 0; e < EXP; e++) if (e != e0 && acc[e] > v1) { v1 = acc[e]; e1 = e; }
    float p1 = expf(v1 - v0);
    float s = 1.f + p1;
    tidx[2 * n] = e0; tidx[2 * n + 1] = e1;
    tp[2 * n] = 1.f / s; tp[2 * n + 1] = p1 / s;
    atomicAdd(&counts[e0], 1);
    atomicAdd(&counts[e1], 1);
  }
}

__global__ void scan_kernel(const int* __restrict__ counts, int* __restrict__ offs) {
  if (threadIdx.x == 0) {
    int s = 0;
    for (int e = 0; e < EXP; e++) { offs[e] = s; s += counts[e]; }
    offs[EXP] = s;
  }
}

__global__ void scatter_kernel(const int* __restrict__ tidx, const float* __restrict__ tp,
                               const int* __restrict__ offs, int* __restrict__ fill,
                               int* __restrict__ token_ids, float* __restrict__ slot_prob) {
  int n = blockIdx.x * blockDim.x + threadIdx.x;
  if (n >= NTOK) return;
#pragma unroll
  for (int j = 0; j < 2; j++) {
    int e = tidx[2 * n + j];
    int slot = offs[e] + atomicAdd(&fill[e], 1);
    token_ids[slot] = n;
    slot_prob[slot] = tp[2 * n + j];
  }
}

// ---------------- GEMM1: H = silu(Xe@W1) * (Xe@W3), gathered A rows ---------
// Double-buffered LDS pipeline: global loads for tile k+1 issue before the
// 8 WMMAs of tile k (L2 latency overlaps matrix work); one barrier per iter.
__global__ __launch_bounds__(256)
void moe_gu_kernel(const unsigned short* __restrict__ Xb,
                   const unsigned short* __restrict__ W1t,   // [E][F][C] bf16
                   const unsigned short* __restrict__ W3t,   // [E][F][C] bf16
                   const int* __restrict__ offs, const int* __restrict__ counts,
                   const int* __restrict__ token_ids,
                   unsigned short* __restrict__ H) {          // [NSLOT][F] bf16
  int e = blockIdx.z, mtile = blockIdx.y, ntile = blockIdx.x;
  int cnt = counts[e];
  if (mtile * BM >= cnt) return;
  int boff = offs[e];

  __shared__ alignas(16) unsigned short As[2 * BM * LDK];
  __shared__ alignas(16) unsigned short B1s[2 * BN * LDK];
  __shared__ alignas(16) unsigned short B3s[2 * BN * LDK];
  __shared__ int toks[BM];

  int tid = threadIdx.x;
  if (tid < BM) {
    int gm = mtile * BM + tid;
    toks[tid] = (gm < cnt) ? token_ids[boff + gm] : -1;
  }
  __syncthreads();

  const unsigned short* W1e = W1t + (size_t)e * FDIM * CDIM + (size_t)(ntile * BN) * CDIM;
  const unsigned short* W3e = W3t + (size_t)e * FDIM * CDIM + (size_t)(ntile * BN) * CDIM;

  int lane = tid & 31, w = tid >> 5;
  int wm = w >> 1, wn = w & 1;                // 4x2 waves -> 32x32 per wave

  v8f accG[2][2] = {};
  v8f accU[2][2] = {};

  int arow = tid >> 1, ahalf = tid & 1;       // 2 threads / A row, 32B each
  int brow = tid >> 2, bq = tid & 3;          // 4 threads / B row, 16B each
  int mytok = toks[arow];

  U4 a0r, a1r, b1r, b3r;                      // prefetch registers
  auto load_tiles = [&](int kt) {
    if (mytok >= 0) {
      const U4* src = (const U4*)(Xb + (size_t)mytok * CDIM + kt + ahalf * 16);
      a0r = src[0]; a1r = src[1];
    } else {
      a0r = U4{0, 0, 0, 0}; a1r = U4{0, 0, 0, 0};
    }
    b1r = *(const U4*)(W1e + (size_t)brow * CDIM + kt + bq * 8);
    b3r = *(const U4*)(W3e + (size_t)brow * CDIM + kt + bq * 8);
  };
  auto store_tiles = [&](int buf) {
    U4* dA = (U4*)(As + buf * (BM * LDK) + arow * LDK + ahalf * 16);
    dA[0] = a0r; dA[1] = a1r;
    *(U4*)(B1s + buf * (BN * LDK) + brow * LDK + bq * 8) = b1r;
    *(U4*)(B3s + buf * (BN * LDK) + brow * LDK + bq * 8) = b3r;
  };

  constexpr int NIT = CDIM / BK;
  load_tiles(0);
  store_tiles(0);
  __syncthreads();

  for (int it = 0; it < NIT; ++it) {
    int buf = it & 1;
    if (it + 1 < NIT) load_tiles((it + 1) * BK);   // overlap L2 latency w/ WMMAs

    const unsigned short* Ab  = As  + buf * (BM * LDK);
    const unsigned short* B1b = B1s + buf * (BN * LDK);
    const unsigned short* B3b = B3s + buf * (BN * LDK);

    v16bf a0  = ld_fragA(Ab,  wm * 32,      lane);
    v16bf a1  = ld_fragA(Ab,  wm * 32 + 16, lane);
    v16bf b10 = ld_fragB(B1b, wn * 32,      lane);
    v16bf b11 = ld_fragB(B1b, wn * 32 + 16, lane);
    v16bf b30 = ld_fragB(B3b, wn * 32,      lane);
    v16bf b31 = ld_fragB(B3b, wn * 32 + 16, lane);

    accG[0][0] = wmma_bf16(a0, b10, accG[0][0]);
    accG[0][1] = wmma_bf16(a0, b11, accG[0][1]);
    accG[1][0] = wmma_bf16(a1, b10, accG[1][0]);
    accG[1][1] = wmma_bf16(a1, b11, accG[1][1]);
    accU[0][0] = wmma_bf16(a0, b30, accU[0][0]);
    accU[0][1] = wmma_bf16(a0, b31, accU[0][1]);
    accU[1][0] = wmma_bf16(a1, b30, accU[1][0]);
    accU[1][1] = wmma_bf16(a1, b31, accU[1][1]);

    if (it + 1 < NIT) store_tiles(buf ^ 1);
    __syncthreads();
  }

  // epilogue: h = silu(g)*u -> bf16 H
  int msub = (lane & 16) ? 8 : 0;
  int ncol = lane & 15;
#pragma unroll
  for (int i = 0; i < 2; i++)
#pragma unroll
    for (int j = 0; j < 2; j++)
#pragma unroll
      for (int r = 0; r < 8; r++) {
        int gm = mtile * BM + wm * 32 + i * 16 + msub + r;
        if (gm < cnt) {
          float g = accG[i][j][r], u = accU[i][j][r];
          float h = (g / (1.f + expf(-g))) * u;
          int f = ntile * BN + wn * 32 + j * 16 + ncol;
          H[(size_t)(boff + gm) * FDIM + f] = f2bf(h);
        }
      }
}

// ---------------- GEMM2: Y = H@W2, weighted atomic scatter to output --------
__global__ __launch_bounds__(256)
void moe_down_kernel(const unsigned short* __restrict__ H,    // [NSLOT][F] bf16
                     const unsigned short* __restrict__ W2t,  // [E][C][F] bf16
                     const int* __restrict__ offs, const int* __restrict__ counts,
                     const int* __restrict__ token_ids, const float* __restrict__ slot_prob,
                     float* __restrict__ out) {
  int e = blockIdx.z, mtile = blockIdx.y, ntile = blockIdx.x;
  int cnt = counts[e];
  if (mtile * BM >= cnt) return;
  int boff = offs[e];

  __shared__ alignas(16) unsigned short As[2 * BM * LDK];
  __shared__ alignas(16) unsigned short Bs[2 * BN * LDK];
  __shared__ int toks[BM];
  __shared__ float probs[BM];

  int tid = threadIdx.x;
  if (tid < BM) {
    int gm = mtile * BM + tid;
    if (gm < cnt) { toks[tid] = token_ids[boff + gm]; probs[tid] = slot_prob[boff + gm]; }
    else          { toks[tid] = -1;                   probs[tid] = 0.f; }
  }
  __syncthreads();

  const unsigned short* W2e = W2t + (size_t)e * CDIM * FDIM + (size_t)(ntile * BN) * FDIM;

  int lane = tid & 31, w = tid >> 5;
  int wm = w >> 1, wn = w & 1;

  v8f accY[2][2] = {};

  int arow = tid >> 1, ahalf = tid & 1;
  int brow = tid >> 2, bq = tid & 3;
  int agm = mtile * BM + arow;
  bool avalid = (agm < cnt);

  U4 a0r, a1r, b0r;
  auto load_tiles = [&](int kt) {
    if (avalid) {
      const U4* src = (const U4*)(H + (size_t)(boff + agm) * FDIM + kt + ahalf * 16);
      a0r = src[0]; a1r = src[1];
    } else {
      a0r = U4{0, 0, 0, 0}; a1r = U4{0, 0, 0, 0};
    }
    b0r = *(const U4*)(W2e + (size_t)brow * FDIM + kt + bq * 8);
  };
  auto store_tiles = [&](int buf) {
    U4* dA = (U4*)(As + buf * (BM * LDK) + arow * LDK + ahalf * 16);
    dA[0] = a0r; dA[1] = a1r;
    *(U4*)(Bs + buf * (BN * LDK) + brow * LDK + bq * 8) = b0r;
  };

  constexpr int NIT = FDIM / BK;
  load_tiles(0);
  store_tiles(0);
  __syncthreads();

  for (int it = 0; it < NIT; ++it) {
    int buf = it & 1;
    if (it + 1 < NIT) load_tiles((it + 1) * BK);

    const unsigned short* Ab = As + buf * (BM * LDK);
    const unsigned short* Bb = Bs + buf * (BN * LDK);

    v16bf a0 = ld_fragA(Ab, wm * 32,      lane);
    v16bf a1 = ld_fragA(Ab, wm * 32 + 16, lane);
    v16bf b0 = ld_fragB(Bb, wn * 32,      lane);
    v16bf b1 = ld_fragB(Bb, wn * 32 + 16, lane);

    accY[0][0] = wmma_bf16(a0, b0, accY[0][0]);
    accY[0][1] = wmma_bf16(a0, b1, accY[0][1]);
    accY[1][0] = wmma_bf16(a1, b0, accY[1][0]);
    accY[1][1] = wmma_bf16(a1, b1, accY[1][1]);

    if (it + 1 < NIT) store_tiles(buf ^ 1);
    __syncthreads();
  }

  // weighted combine: exactly 2 commutative fp32 adds per output element -> deterministic
  int msub = (lane & 16) ? 8 : 0;
  int ncol = lane & 15;
#pragma unroll
  for (int i = 0; i < 2; i++)
#pragma unroll
    for (int j = 0; j < 2; j++)
#pragma unroll
      for (int r = 0; r < 8; r++) {
        int lrow = wm * 32 + i * 16 + msub + r;
        int gm = mtile * BM + lrow;
        if (gm < cnt) {
          int tok = toks[lrow];
          float p = probs[lrow];
          int c = ntile * BN + wn * 32 + j * 16 + ncol;
          unsafeAtomicAdd(out + (size_t)tok * CDIM + c, p * accY[i][j][r]);
        }
      }
}

// ---------------- host launcher ----------------
extern "C" void kernel_launch(void* const* d_in, const int* in_sizes, int n_in,
                              void* d_out, int out_size, void* d_ws, size_t ws_size,
                              hipStream_t stream) {
  const float* x  = (const float*)d_in[0];   // [N, C]
  const float* wr = (const float*)d_in[1];   // [C, E]
  const float* w1 = (const float*)d_in[2];   // [E, C, F]
  const float* w3 = (const float*)d_in[3];   // [E, C, F]
  const float* w2 = (const float*)d_in[4];   // [E, F, C]
  float* out = (float*)d_out;

  // workspace layout (256B aligned regions), total ~133 MB
  char* ws = (char*)d_ws;
  size_t o = 0;
  auto carve = [&](size_t bytes) { size_t r = o; o = (o + bytes + 255) & ~(size_t)255; return r; };
  size_t o_cnt  = carve(2 * EXP * sizeof(int));            // counts[8] + fill[8]
  size_t o_offs = carve((EXP + 1) * sizeof(int));
  size_t o_tidx = carve((size_t)2 * NTOK * sizeof(int));
  size_t o_tp   = carve((size_t)2 * NTOK * sizeof(float));
  size_t o_tok  = carve((size_t)NSLOT * sizeof(int));
  size_t o_sp   = carve((size_t)NSLOT * sizeof(float));
  size_t o_xb   = carve((size_t)NTOK * CDIM * 2);
  size_t o_w1t  = carve((size_t)EXP * FDIM * CDIM * 2);
  size_t o_w3t  = carve((size_t)EXP * FDIM * CDIM * 2);
  size_t o_w2t  = carve((size_t)EXP * CDIM * FDIM * 2);
  size_t o_h    = carve((size_t)NSLOT * FDIM * 2);
  (void)ws_size; (void)in_sizes; (void)n_in;

  int*            counts = (int*)(ws + o_cnt);
  int*            fill   = counts + EXP;
  int*            offs   = (int*)(ws + o_offs);
  int*            tidx   = (int*)(ws + o_tidx);
  float*          tp     = (float*)(ws + o_tp);
  int*            toks   = (int*)(ws + o_tok);
  float*          sprob  = (float*)(ws + o_sp);
  unsigned short* Xb     = (unsigned short*)(ws + o_xb);
  unsigned short* W1t    = (unsigned short*)(ws + o_w1t);
  unsigned short* W3t    = (unsigned short*)(ws + o_w3t);
  unsigned short* W2t    = (unsigned short*)(ws + o_w2t);
  unsigned short* H      = (unsigned short*)(ws + o_h);

  hipMemsetAsync(counts, 0, 2 * EXP * sizeof(int), stream);
  hipMemsetAsync(out, 0, (size_t)out_size * sizeof(float), stream);

  // bf16 conversions (weights end up L2-resident: ~100MB bf16 << 192MB L2)
  {
    int n = NTOK * CDIM;
    cvt_x_kernel<<<(n + 255) / 256, 256, 0, stream>>>(x, Xb, n);
  }
  transpose_bf16_kernel<<<dim3(FDIM / 32, CDIM / 32, EXP), dim3(32, 8), 0, stream>>>(w1, W1t, CDIM, FDIM);
  transpose_bf16_kernel<<<dim3(FDIM / 32, CDIM / 32, EXP), dim3(32, 8), 0, stream>>>(w3, W3t, CDIM, FDIM);
  transpose_bf16_kernel<<<dim3(CDIM / 32, FDIM / 32, EXP), dim3(32, 8), 0, stream>>>(w2, W2t, FDIM, CDIM);

  // routing (pure fp32: selection must match reference exactly)
  router_kernel<<<NTOK / 8, 256, 0, stream>>>(x, wr, counts, tidx, tp);
  scan_kernel<<<1, 32, 0, stream>>>(counts, offs);
  scatter_kernel<<<NTOK / 256, 256, 0, stream>>>(tidx, tp, offs, fill, toks, sprob);

  // expert GEMMs (bf16 WMMA, fp32 accumulate)
  moe_gu_kernel<<<dim3(FDIM / BN, NTOK / BM, EXP), 256, 0, stream>>>(
      Xb, W1t, W3t, offs, counts, toks, H);
  moe_down_kernel<<<dim3(CDIM / BN, NTOK / BM, EXP), 256, 0, stream>>>(
      H, W2t, offs, counts, toks, sprob, out);
}
